// MoEBlock_919123001779
// MI455X (gfx1250) — compile-verified, matrix operand order checked by reading, not run on hardware
//
#include <hip/hip_runtime.h>
#include <hip/hip_bf16.h>
#include <stdint.h>

// ---------------- problem constants ----------------
#define NTOK   131072
#define DIM    256
#define HID    512
#define NEXP   16
#define TOPK   2
#define CAP    18023            // ceil(N*K/E * 1.1)
#define MT     64               // rows per expert tile (4 sub-tiles of 16)
#define RS     4                // row sub-tiles per block
#define NT1    (HID/16)         // 32 n-tiles in GEMM1
#define KS1    (DIM/32)         // 8  k-steps  in GEMM1
#define NT2    (DIM/16)         // 16 n-tiles in GEMM2
#define KS2    (HID/32)         // 16 k-steps  in GEMM2
#define W1ELEM (NEXP*DIM*HID)   // 2,097,152
#define W2ELEM (NEXP*HID*DIM)

typedef __attribute__((ext_vector_type(16))) __bf16 v16bf;
typedef __attribute__((ext_vector_type(8)))  float  v8f;

__device__ __forceinline__ __bf16 f2bf(float f) {
    unsigned u = __builtin_bit_cast(unsigned, f);
    u += 0x7fffu + ((u >> 16) & 1u);             // round-to-nearest-even
    unsigned short h = (unsigned short)(u >> 16);
    return __builtin_bit_cast(__bf16, h);
}

// ---------------- zero fill (graph-capture safe, no memset) ----------------
__global__ void moe_zero_kernel(float* __restrict__ p, long long n) {
    long long i = (long long)blockIdx.x * blockDim.x + threadIdx.x;
    long long stride = (long long)gridDim.x * blockDim.x;
    for (; i < n; i += stride) p[i] = 0.0f;
}

// ---------------- weight convert + swizzle into WMMA fragment layout -------
// dst index = (((e*NT + nt)*KS + ks)*32 + lane)*16 + i
// lane fragment element i -> (k, n):
//   part=i>>3, j=i&7, half=lane>>4, col=lane&15
//   k = ks*32 + part*16 + half*8 + j ;  n = nt*16 + col
__global__ __launch_bounds__(256) void moe_convert_w_kernel(
    const float* __restrict__ W1, const float* __restrict__ W2,
    __bf16* __restrict__ W1f, __bf16* __restrict__ W2f)
{
    int idx = blockIdx.x * blockDim.x + threadIdx.x;   // [0, W1ELEM + W2ELEM)
    if (idx < W1ELEM) {
        int i = idx & 15, lane = (idx >> 4) & 31, r = idx >> 9;
        int ks = r & (KS1 - 1); r >>= 3;
        int nt = r & (NT1 - 1); int e = r >> 5;
        int part = i >> 3, j = i & 7, half = lane >> 4, col = lane & 15;
        int k = ks * 32 + part * 16 + half * 8 + j;
        int n = nt * 16 + col;
        W1f[idx] = f2bf(W1[((size_t)e * DIM + k) * HID + n]);
    } else {
        int t = idx - W1ELEM;
        int i = t & 15, lane = (t >> 4) & 31, r = t >> 9;
        int ks = r & (KS2 - 1); r >>= 4;
        int nt = r & (NT2 - 1); int e = r >> 4;
        int part = i >> 3, j = i & 7, half = lane >> 4, col = lane & 15;
        int k = ks * 32 + part * 16 + half * 8 + j;
        int n = nt * 16 + col;
        W2f[t] = f2bf(W2[((size_t)e * HID + k) * DIM + n]);
    }
}

// ---------------- router + top-2 dispatch (one wave per token) -------------
__global__ __launch_bounds__(256) void moe_router_kernel(
    const float* __restrict__ ctx, const float* __restrict__ Wg,
    const float* __restrict__ bg,
    int* __restrict__ counts, int* __restrict__ tok_tab, float* __restrict__ sc_tab,
    float* __restrict__ imp_acc, float* __restrict__ ent_acc)
{
    __shared__ float sh_imp[NEXP];
    __shared__ float sh_ent;
    int tid = threadIdx.x, lane = tid & 31, wave = tid >> 5;
    if (tid < NEXP) sh_imp[tid] = 0.0f;
    if (tid == 0) sh_ent = 0.0f;
    __syncthreads();

    int tok = blockIdx.x * 8 + wave;

    float acc[NEXP];
    #pragma unroll
    for (int e = 0; e < NEXP; ++e) acc[e] = 0.0f;

    const float* x = ctx + (size_t)tok * DIM + lane * 8;
    #pragma unroll
    for (int j = 0; j < 8; ++j) {
        float xv = x[j];
        const float* wrow = Wg + (size_t)(lane * 8 + j) * NEXP;
        #pragma unroll
        for (int e = 0; e < NEXP; ++e) acc[e] = fmaf(xv, wrow[e], acc[e]);
    }
    // butterfly reduce: every lane ends with the full 16 logits
    #pragma unroll
    for (int off = 16; off > 0; off >>= 1) {
        #pragma unroll
        for (int e = 0; e < NEXP; ++e) acc[e] += __shfl_xor(acc[e], off);
    }
    #pragma unroll
    for (int e = 0; e < NEXP; ++e) acc[e] += bg[e];

    // softmax stats (TEMP = 1)
    float mx = acc[0];
    #pragma unroll
    for (int e = 1; e < NEXP; ++e) mx = fmaxf(mx, acc[e]);
    float ex[NEXP]; float sum = 0.0f;
    #pragma unroll
    for (int e = 0; e < NEXP; ++e) { ex[e] = __expf(acc[e] - mx); sum += ex[e]; }
    float inv = 1.0f / sum;
    float lse = __logf(sum);

    if (lane < NEXP) atomicAdd(&sh_imp[lane], ex[lane] * inv);

    if (lane == 0) {
        float ent = 0.0f;
        #pragma unroll
        for (int e = 0; e < NEXP; ++e) {
            float p = ex[e] * inv;
            ent -= p * (acc[e] - mx - lse);       // p * log p
        }
        atomicAdd(&sh_ent, ent);

        // top-2 (first occurrence wins ties, like top_k)
        int i1 = 0; float v1 = acc[0];
        #pragma unroll
        for (int e = 1; e < NEXP; ++e) if (acc[e] > v1) { v1 = acc[e]; i1 = e; }
        int i2 = (i1 == 0) ? 1 : 0; float v2 = acc[i2];
        #pragma unroll
        for (int e = 0; e < NEXP; ++e)
            if (e != i1 && acc[e] > v2) { v2 = acc[e]; i2 = e; }
        float g1 = 1.0f / (1.0f + __expf(v2 - v1));
        float g2 = 1.0f - g1;

        int r = atomicAdd(&counts[i1], 1);
        if (r < CAP) { tok_tab[i1 * CAP + r] = tok; sc_tab[i1 * CAP + r] = g1; }
        r = atomicAdd(&counts[i2], 1);
        if (r < CAP) { tok_tab[i2 * CAP + r] = tok; sc_tab[i2 * CAP + r] = g2; }
    }
    __syncthreads();
    if (tid < NEXP) atomicAdd(&imp_acc[tid], sh_imp[tid]);
    if (tid == 0)   atomicAdd(ent_acc, sh_ent);
}

// ---------------- fused expert MLP: relu(X@W1+b1)@W2+b2, scaled scatter ----
// 64-row tile per block; each B fragment is loaded once from L2 and reused
// across 4 row-subtile WMMAs (A fragments re-read from LDS).
__global__ __launch_bounds__(256) void moe_expert_kernel(
    const float* __restrict__ ctx,
    const __bf16* __restrict__ W1f, const __bf16* __restrict__ W2f,
    const float* __restrict__ b1,  const float* __restrict__ b2,
    const int* __restrict__ counts, const int* __restrict__ tok_tab,
    const float* __restrict__ sc_tab, float* __restrict__ out)
{
    const int tiles = (CAP + MT - 1) / MT;           // 282
    int e    = blockIdx.x / tiles;
    int tile = blockIdx.x % tiles;
    int cnt  = counts[e]; if (cnt > CAP) cnt = CAP;
    int row0 = tile * MT;
    if (row0 >= cnt) return;

    __shared__ __bf16 Xs[MT][DIM + 8];
    __shared__ __bf16 Hs[MT][HID + 8];
    __shared__ int    toks[MT];
    __shared__ float  scs[MT];

    int tid = threadIdx.x, wave = tid >> 5, lane = tid & 31;
    int half = lane >> 4, hl8 = half * 8, rowA = lane & 15, col = lane & 15;

    if (tid < MT) {
        int r = row0 + tid; int tok = 0; float sc = 0.0f;
        if (r < cnt) { tok = tok_tab[e * CAP + r]; sc = sc_tab[e * CAP + r]; }
        toks[tid] = tok; scs[tid] = sc;
    }
    __syncthreads();

    // gather + convert X tile [64 x 256] -> LDS bf16 (coalesced: 16 lanes/row)
    {
        int rbase = tid >> 4;            // 0..15
        int seg   = (tid & 15) * 16;
        #pragma unroll
        for (int rr = 0; rr < RS; ++rr) {
            int row = rbase + rr * 16;
            const float* src = ctx + (size_t)toks[row] * DIM + seg;
            #pragma unroll
            for (int j = 0; j < 16; ++j) Xs[row][seg + j] = f2bf(src[j]);
        }
    }
    __syncthreads();

    // ---- GEMM1: H = relu(X @ W1 + b1); wave owns 4 n-tiles x 4 row-subtiles
    #pragma unroll
    for (int t = 0; t < 4; ++t) {
        int nt = wave * 4 + t;
        v8f acc[RS] = {};
        #pragma unroll
        for (int ks = 0; ks < KS1; ++ks) {
            const __bf16* bp = W1f + (size_t)((((e * NT1 + nt) * KS1) + ks) * 32 + lane) * 16;
            v16bf b;
            #pragma unroll
            for (int j = 0; j < 16; ++j) b[j] = bp[j];
            int kb = ks * 32;
            #pragma unroll
            for (int rs = 0; rs < RS; ++rs) {
                int row = rs * 16 + rowA;
                v16bf a;
                #pragma unroll
                for (int j = 0; j < 8; ++j) {
                    a[j]     = Xs[row][kb + hl8 + j];
                    a[8 + j] = Xs[row][kb + 16 + hl8 + j];
                }
                acc[rs] = __builtin_amdgcn_wmma_f32_16x16x32_bf16(
                              false, a, false, b, (short)0, acc[rs], false, false);
            }
        }
        int n = nt * 16 + col;
        float bias = b1[e * HID + n];
        #pragma unroll
        for (int rs = 0; rs < RS; ++rs) {
            #pragma unroll
            for (int i = 0; i < 8; ++i) {
                float v = acc[rs][i] + bias;
                v = v > 0.0f ? v : 0.0f;
                Hs[rs * 16 + i + half * 8][n] = f2bf(v);
            }
        }
    }
    __syncthreads();

    // ---- GEMM2: Y = H @ W2 + b2; wave owns 2 n-tiles x 4 row-subtiles ----
    #pragma unroll
    for (int t = 0; t < 2; ++t) {
        int nt = wave * 2 + t;
        v8f acc[RS] = {};
        #pragma unroll
        for (int ks = 0; ks < KS2; ++ks) {
            const __bf16* bp = W2f + (size_t)((((e * NT2 + nt) * KS2) + ks) * 32 + lane) * 16;
            v16bf b;
            #pragma unroll
            for (int j = 0; j < 16; ++j) b[j] = bp[j];
            int kb = ks * 32;
            #pragma unroll
            for (int rs = 0; rs < RS; ++rs) {
                int row = rs * 16 + rowA;
                v16bf a;
                #pragma unroll
                for (int j = 0; j < 8; ++j) {
                    a[j]     = Hs[row][kb + hl8 + j];
                    a[8 + j] = Hs[row][kb + 16 + hl8 + j];
                }
                acc[rs] = __builtin_amdgcn_wmma_f32_16x16x32_bf16(
                              false, a, false, b, (short)0, acc[rs], false, false);
            }
        }
        int n = nt * 16 + col;
        float bias = b2[e * DIM + n];
        #pragma unroll
        for (int rs = 0; rs < RS; ++rs) {
            #pragma unroll
            for (int i = 0; i < 8; ++i) {
                int m = rs * 16 + i + half * 8;
                float sc = scs[m];
                if (sc != 0.0f) {
                    float v = (acc[rs][i] + bias) * sc;
                    atomicAdd(&out[(size_t)toks[m] * DIM + n], v);
                }
            }
        }
    }
}

// ---------------- aux stats ----------------
__global__ void moe_finalize_kernel(const int* __restrict__ counts,
                                    const float* __restrict__ imp,
                                    const float* __restrict__ ent,
                                    float* __restrict__ out_tail)
{
    __shared__ float disp[NEXP];
    int t = threadIdx.x;
    if (t < NEXP) {
        int c = counts[t]; if (c > CAP) c = CAP;
        disp[t] = (float)c;
    }
    __syncthreads();
    if (t == 0) {
        float tot = 0.0f;
        for (int e = 0; e < NEXP; ++e) tot += disp[e];
        float denom = fmaxf(tot, 1.0f);
        float aux = 0.0f;
        for (int e = 0; e < NEXP; ++e)
            aux += (imp[e] / (float)NTOK) * (disp[e] / denom);
        out_tail[0] = aux * (float)NEXP;          // aux_loss
        out_tail[1] = ent[0] / (float)NTOK;       // entropy
        for (int e = 0; e < NEXP; ++e) out_tail[2 + e] = disp[e];
    }
}

// ---------------- launch ----------------
extern "C" void kernel_launch(void* const* d_in, const int* in_sizes, int n_in,
                              void* d_out, int out_size, void* d_ws, size_t ws_size,
                              hipStream_t stream)
{
    const float* ctx = (const float*)d_in[0];
    const float* Wg  = (const float*)d_in[1];
    const float* bg  = (const float*)d_in[2];
    const float* W1  = (const float*)d_in[3];
    const float* b1  = (const float*)d_in[4];
    const float* W2  = (const float*)d_in[5];
    const float* b2  = (const float*)d_in[6];
    float* out = (float*)d_out;

    // workspace layout (~10.7 MB)
    char* ws = (char*)d_ws;
    int*    counts  = (int*)(ws + 0);
    float*  imp     = (float*)(ws + 64);
    float*  ent     = (float*)(ws + 128);
    int*    tok_tab = (int*)(ws + 256);
    float*  sc_tab  = (float*)(ws + 256 + (size_t)NEXP * CAP * 4);
    __bf16* W1f     = (__bf16*)(ws + 256 + 2 * (size_t)NEXP * CAP * 4);
    __bf16* W2f     = W1f + W1ELEM;

    // zero: output (scatter-add target) + ws header/tables
    long long outN = (long long)NTOK * DIM + 2 + NEXP;   // full d_out
    moe_zero_kernel<<<4096, 256, 0, stream>>>(out, outN);
    long long wsN = (256 + 2LL * NEXP * CAP * 4) / 4;    // counts..sc_tab (as f32 words)
    moe_zero_kernel<<<1024, 256, 0, stream>>>((float*)ws, wsN);

    // weight convert + fragment swizzle (bf16)
    moe_convert_w_kernel<<<(W1ELEM + W2ELEM) / 256, 256, 0, stream>>>(W1, W2, W1f, W2f);

    // router + dispatch
    moe_router_kernel<<<NTOK / 8, 256, 0, stream>>>(ctx, Wg, bg, counts, tok_tab,
                                                    sc_tab, imp, ent);

    // fused expert MLP with WMMA (64-row tiles, 4x B-fragment reuse)
    int tiles = (CAP + MT - 1) / MT;                     // 282
    moe_expert_kernel<<<NEXP * tiles, 256, 0, stream>>>(ctx, W1f, W2f, b1, b2,
                                                        counts, tok_tab, sc_tab, out);

    // aux outputs
    moe_finalize_kernel<<<1, 32, 0, stream>>>(counts, imp, ent, out + (size_t)NTOK * DIM);
}
